// AttentionUnit_28020366639361
// MI455X (gfx1250) — compile-verified, matrix operand order checked by reading, not run on hardware
//
#include <hip/hip_runtime.h>

#define BATCH 4
#define SEQ   2048
#define DM    1024
#define NH    16
#define HD    64
#define M_TOT (BATCH * SEQ)   // 8192

typedef __attribute__((ext_vector_type(16))) _Float16 v16h;
typedef __attribute__((ext_vector_type(8)))  _Float16 v8h;
typedef __attribute__((ext_vector_type(4)))  _Float16 v4h;
typedef __attribute__((ext_vector_type(8)))  float    v8f;
typedef __attribute__((ext_vector_type(4)))  int      v4i;

// ---- gfx1250 async global->LDS copy (ASYNCcnt path), with safe fallback ----
#if defined(__has_builtin)
# if __has_builtin(__builtin_amdgcn_global_load_async_to_lds_b128)
#  define HAVE_ASYNC_COPY 1
# endif
#endif
#ifndef HAVE_ASYNC_COPY
# define HAVE_ASYNC_COPY 0
#endif

__device__ __forceinline__ void async_cp16(void* lds_dst, const void* gsrc) {
#if HAVE_ASYNC_COPY
  __builtin_amdgcn_global_load_async_to_lds_b128((v4i*)gsrc, (v4i*)lds_dst, 0,
                                                 0);
#else
  *(v8h*)lds_dst = *(const v8h*)gsrc;
#endif
}

__device__ __forceinline__ void async_wait0() {
#if HAVE_ASYNC_COPY
# if defined(__has_builtin) && __has_builtin(__builtin_amdgcn_s_wait_asynccnt)
  __builtin_amdgcn_s_wait_asynccnt(0);
# else
  asm volatile("s_wait_asynccnt 0x0" ::: "memory");
# endif
#endif
}

__device__ __forceinline__ v8f zero8() {
  v8f z;
#pragma unroll
  for (int i = 0; i < 8; ++i) z[i] = 0.0f;
  return z;
}

// Load one 16x32 f16 WMMA fragment (A layout; B uses the same pattern when the
// matrix is stored column-of-output-major). Lane L: row = L&15, the two 16-byte
// chunks cover k = (L>>4)*8..+7 and k = 16+(L>>4)*8..+7. stride in f16 elems.
__device__ __forceinline__ v16h frag_ld(const _Float16* base, int stride) {
  const int lane = threadIdx.x & 31;
  const int r    = lane & 15;
  const int hi   = lane >> 4;
  const _Float16* p = base + r * stride + hi * 8;
  v8h a = *(const v8h*)(p);
  v8h b = *(const v8h*)(p + 16);
  return __builtin_shufflevector(a, b, 0,1,2,3,4,5,6,7,8,9,10,11,12,13,14,15);
}

__device__ __forceinline__ v8f wmma16(v16h a, v16h b, v8f c) {
  return __builtin_amdgcn_wmma_f32_16x16x32_f16(false, a, false, b, (short)0, c,
                                                false, false);
}

// ---------------------------------------------------------------------------
// GEMM: C[M,N] = A[M,1024] @ W[1024,N] + bias, M=8192, N=1024.
// Block tile 128x64, BK=64 (8 WMMA / stage), 256 threads = 8 waves (4x2).
// MODE 0: Q  -> f16 [B,H,S,64], scaled by 1/sqrt(64)
// MODE 1: K  -> f16 [B,H,S,64]
// MODE 2: V  -> f16 [B,H,64,S]  (transposed per head)
// MODE 3: A is f16 (attention output), result f32 -> d_out
// ---------------------------------------------------------------------------
template <int MODE>
__global__ __launch_bounds__(256) void proj_kernel(
    const float* __restrict__ Af, const _Float16* __restrict__ Ah,
    const float* __restrict__ W, const float* __restrict__ bias,
    _Float16* __restrict__ outh, float* __restrict__ outf) {
  constexpr int BK = 64;
  constexpr int LD = BK + 8;  // 72 halves = 144B row stride (16B aligned)
  __shared__ _Float16 As[128][LD];
  __shared__ _Float16 Bs[64][LD];

  const int tid  = threadIdx.x;
  const int lane = tid & 31;
  const int wid  = tid >> 5;
  const int wm   = wid & 3;   // 0..3 (M)
  const int wn   = wid >> 2;  // 0..1 (N)
  const int bm0  = blockIdx.x * 128;
  const int bn0  = blockIdx.y * 64;

  v8f c00 = zero8(), c01 = zero8(), c10 = zero8(), c11 = zero8();

  for (int k0 = 0; k0 < DM; k0 += BK) {
    __syncthreads();
    if constexpr (MODE < 3) {
      // 128x64 f32 -> f16, packed 8-byte LDS stores
      const int r0 = tid >> 4;           // 0..15
      const int c4 = (tid & 15) * 4;     // 0..60
#pragma unroll
      for (int rr = 0; rr < 8; ++rr) {
        const int row = r0 + rr * 16;
        const float4 v =
            *(const float4*)(Af + (size_t)(bm0 + row) * DM + k0 + c4);
        v4h h;
        h[0] = (_Float16)v.x; h[1] = (_Float16)v.y;
        h[2] = (_Float16)v.z; h[3] = (_Float16)v.w;
        *(v4h*)&As[row][c4] = h;
      }
    } else {
      // straight f16 copy -> async to LDS
      const int row = tid >> 1;          // 0..127
      const int c0  = (tid & 1) * 32;
      const _Float16* src = Ah + (size_t)(bm0 + row) * DM + k0 + c0;
#pragma unroll
      for (int i = 0; i < 4; ++i)
        async_cp16(&As[row][c0 + i * 8], src + i * 8);
    }
    {  // W tile, stored transposed: Bs[n][k]
      const int k  = tid >> 4;           // 0..15
      const int n4 = (tid & 15) * 4;
#pragma unroll
      for (int kk = 0; kk < 4; ++kk) {
        const int krow = k + kk * 16;
        const float4 v =
            *(const float4*)(W + (size_t)(k0 + krow) * DM + bn0 + n4);
        Bs[n4 + 0][krow] = (_Float16)v.x;
        Bs[n4 + 1][krow] = (_Float16)v.y;
        Bs[n4 + 2][krow] = (_Float16)v.z;
        Bs[n4 + 3][krow] = (_Float16)v.w;
      }
    }
    if constexpr (MODE == 3) async_wait0();
    __syncthreads();

#pragma unroll
    for (int ks = 0; ks < 2; ++ks) {
      v16h a0 = frag_ld(&As[wm * 32][ks * 32], LD);
      v16h a1 = frag_ld(&As[wm * 32 + 16][ks * 32], LD);
      v16h b0 = frag_ld(&Bs[wn * 32][ks * 32], LD);
      v16h b1 = frag_ld(&Bs[wn * 32 + 16][ks * 32], LD);
      c00 = wmma16(a0, b0, c00);
      c01 = wmma16(a0, b1, c01);
      c10 = wmma16(a1, b0, c10);
      c11 = wmma16(a1, b1, c11);
    }
  }

  const int hi  = lane >> 4;
  const int col = lane & 15;
  v8f acc[2][2] = {{c00, c01}, {c10, c11}};
#pragma unroll
  for (int i = 0; i < 2; ++i) {
#pragma unroll
    for (int j = 0; j < 2; ++j) {
      const int gn   = bn0 + wn * 32 + j * 16 + col;
      const float bj = bias[gn];
#pragma unroll
      for (int v = 0; v < 8; ++v) {
        const int gm = bm0 + wm * 32 + i * 16 + v + 8 * hi;
        float val    = acc[i][j][v] + bj;
        if constexpr (MODE == 3) {
          outf[(size_t)gm * DM + gn] = val;
        } else {
          const int b = gm >> 11, s = gm & (SEQ - 1);
          const int h = gn >> 6, dh = gn & 63;
          if constexpr (MODE == 0) val *= 0.125f;  // 1/sqrt(HEAD_DIM)
          size_t off;
          if constexpr (MODE == 2)
            off = ((size_t)(b * NH + h) * HD + dh) * SEQ + s;
          else
            off = ((size_t)(b * NH + h) * SEQ + s) * HD + dh;
          outh[off] = (_Float16)val;
        }
      }
    }
  }
}

// ---------------------------------------------------------------------------
// Flash attention: block = (b,h, 64 query rows), 4 waves x 16 rows.
// ---------------------------------------------------------------------------
__global__ __launch_bounds__(128) void attn_kernel(
    const _Float16* __restrict__ Q, const _Float16* __restrict__ K,
    const _Float16* __restrict__ Vt, const unsigned char* __restrict__ mask,
    _Float16* __restrict__ Aout) {
  __shared__ _Float16 Ks[64][72];       // [j][d]
  __shared__ _Float16 Vs[64][72];       // [d][j]
  __shared__ _Float16 Ps[4][16][72];    // per-wave P tile [row][j]

  const int tid   = threadIdx.x;
  const int lane  = tid & 31;
  const int w     = tid >> 5;
  const int col   = lane & 15;
  const int hi    = lane >> 4;
  const int bh    = blockIdx.x;
  const int b     = bh >> 4;
  const int h     = bh & 15;
  const int q0    = blockIdx.y * 64;
  const int qrow0 = q0 + w * 16;

  const _Float16* Qbase = Q + ((size_t)bh * SEQ + qrow0) * HD;
  v16h qa0 = frag_ld(Qbase, HD);        // d = 0..31
  v16h qa1 = frag_ld(Qbase + 32, HD);   // d = 32..63

  v8f o0 = zero8(), o1 = zero8(), o2 = zero8(), o3 = zero8();
  float mrow[8], lrow[8];
#pragma unroll
  for (int v = 0; v < 8; ++v) { mrow[v] = -1e30f; lrow[v] = 0.0f; }

  const int row = tid >> 1;             // 0..63
  const int c0  = (tid & 1) * 32;

  for (int j0 = 0; j0 <= q0; j0 += 64) {           // causal chunk pruning
    if (mask[b * SEQ + j0]) break;                  // monotonic tail padding
    __syncthreads();
    {
      const _Float16* ks = K + ((size_t)bh * SEQ + j0 + row) * HD + c0;
      const _Float16* vs = Vt + ((size_t)bh * HD + row) * SEQ + j0 + c0;
#pragma unroll
      for (int i = 0; i < 4; ++i) {
        async_cp16(&Ks[row][c0 + i * 8], ks + i * 8);
        async_cp16(&Vs[row][c0 + i * 8], vs + i * 8);
      }
      async_wait0();
    }
    __syncthreads();

    // S = Q * K^T  (per wave: 16 rows x 64 cols = 4 tiles, 2 WMMA each)
    v8f st[4];
#pragma unroll
    for (int u = 0; u < 4; ++u) {
      v16h kb0 = frag_ld(&Ks[u * 16][0], 72);
      v16h kb1 = frag_ld(&Ks[u * 16][32], 72);
      v8f s = zero8();
      s = wmma16(qa0, kb0, s);
      s = wmma16(qa1, kb1, s);
      st[u] = s;
    }

    // causal + key-padding masks
#pragma unroll
    for (int u = 0; u < 4; ++u) {
      const int gj   = j0 + u * 16 + col;
      const bool pad = mask[b * SEQ + gj] != 0;
#pragma unroll
      for (int v = 0; v < 8; ++v) {
        const int gi = qrow0 + v + 8 * hi;
        float sv = st[u][v];
        if (pad || gj > gi) sv = -1e30f;
        st[u][v] = sv;
      }
    }

    // online softmax (rows live across 16-lane halves in C layout)
#pragma unroll
    for (int v = 0; v < 8; ++v) {
      float mloc = fmaxf(fmaxf(st[0][v], st[1][v]), fmaxf(st[2][v], st[3][v]));
#pragma unroll
      for (int x = 1; x < 16; x <<= 1)
        mloc = fmaxf(mloc, __shfl_xor(mloc, x, 32));
      const float mnew  = fmaxf(mrow[v], mloc);
      const float scale = __expf(mrow[v] - mnew);
      mrow[v] = mnew;
      float rsum = 0.0f;
#pragma unroll
      for (int u = 0; u < 4; ++u) {
        const float p = __expf(st[u][v] - mnew);
        st[u][v] = p;
        rsum += p;
      }
#pragma unroll
      for (int x = 1; x < 16; x <<= 1) rsum += __shfl_xor(rsum, x, 32);
      lrow[v] = lrow[v] * scale + rsum;
      o0[v] *= scale; o1[v] *= scale; o2[v] *= scale; o3[v] *= scale;
    }

    // C-layout -> A-layout via per-wave LDS (in-order within wave)
#pragma unroll
    for (int u = 0; u < 4; ++u)
#pragma unroll
      for (int v = 0; v < 8; ++v)
        Ps[w][v + 8 * hi][u * 16 + col] = (_Float16)st[u][v];

    // O += P * V
#pragma unroll
    for (int kk = 0; kk < 2; ++kk) {
      v16h pa  = frag_ld(&Ps[w][0][kk * 32], 72);
      v16h vb0 = frag_ld(&Vs[0][kk * 32], 72);
      v16h vb1 = frag_ld(&Vs[16][kk * 32], 72);
      v16h vb2 = frag_ld(&Vs[32][kk * 32], 72);
      v16h vb3 = frag_ld(&Vs[48][kk * 32], 72);
      o0 = wmma16(pa, vb0, o0);
      o1 = wmma16(pa, vb1, o1);
      o2 = wmma16(pa, vb2, o2);
      o3 = wmma16(pa, vb3, o3);
    }
  }

  // normalize, reassemble [B,S,H*64] f16 for the output projection
#pragma unroll
  for (int v = 0; v < 8; ++v) {
    const float inv = lrow[v] > 0.0f ? 1.0f / lrow[v] : 0.0f;
    const int gi    = qrow0 + v + 8 * hi;
    const size_t base = ((size_t)b * SEQ + gi) * DM + h * HD;
    Aout[base + 0  + col] = (_Float16)(o0[v] * inv);
    Aout[base + 16 + col] = (_Float16)(o1[v] * inv);
    Aout[base + 32 + col] = (_Float16)(o2[v] * inv);
    Aout[base + 48 + col] = (_Float16)(o3[v] * inv);
  }
}

extern "C" void kernel_launch(void* const* d_in, const int* in_sizes, int n_in,
                              void* d_out, int out_size, void* d_ws,
                              size_t ws_size, hipStream_t stream) {
  const float* q_in = (const float*)d_in[0];
  const float* k_in = (const float*)d_in[1];
  const float* v_in = (const float*)d_in[2];
  const unsigned char* mask = (const unsigned char*)d_in[3];
  const float* Wq = (const float*)d_in[4];
  const float* bq = (const float*)d_in[5];
  const float* Wk = (const float*)d_in[6];
  const float* bk = (const float*)d_in[7];
  const float* Wv = (const float*)d_in[8];
  const float* bv = (const float*)d_in[9];
  const float* Wo = (const float*)d_in[10];
  const float* bo = (const float*)d_in[11];

  char* ws = (char*)d_ws;
  const size_t seg = (size_t)M_TOT * DM * sizeof(_Float16);  // 16 MiB each
  _Float16* Qh  = (_Float16*)(ws + 0 * seg);
  _Float16* Kh  = (_Float16*)(ws + 1 * seg);
  _Float16* Vth = (_Float16*)(ws + 2 * seg);
  _Float16* Ah  = (_Float16*)(ws + 3 * seg);

  dim3 gp(M_TOT / 128, DM / 64);
  proj_kernel<0><<<gp, 256, 0, stream>>>(q_in, nullptr, Wq, bq, Qh, nullptr);
  proj_kernel<1><<<gp, 256, 0, stream>>>(k_in, nullptr, Wk, bk, Kh, nullptr);
  proj_kernel<2><<<gp, 256, 0, stream>>>(v_in, nullptr, Wv, bv, Vth, nullptr);
  attn_kernel<<<dim3(BATCH * NH, SEQ / 64), 128, 0, stream>>>(Qh, Kh, Vth, mask,
                                                              Ah);
  proj_kernel<3><<<gp, 256, 0, stream>>>(nullptr, Ah, Wo, bo, nullptr,
                                         (float*)d_out);
}
